// EGTLayer_60129542685
// MI455X (gfx1250) — matrix operand order repaired and checked by reading, not compiled, not run on hardware
//
#include <hip/hip_runtime.h>
#include <math.h>

// ---------------------------------------------------------------------------
// EGT layer for gfx1250 (MI455X). All GEMMs use v_wmma_f32_16x16x32_bf16
// (bf16 inputs, f32 accumulate). Edge-side e_out + edge-FFN fully fused.
// Fragment loaders: compile-time column stride; CS==1 uses float4 loads.
// All launches tile exactly -> no bounds checks, no divergence in hot loops.
// GEMM: block = 4 waves stacked in M, wave tile 32x64 (2x4 accumulators).
// ---------------------------------------------------------------------------

typedef __attribute__((ext_vector_type(16))) __bf16 v16bf;
typedef __attribute__((ext_vector_type(8)))  float  v8f;

#define FLAG_ELU  1
#define FLAG_CLIP 2

__device__ __forceinline__ v8f vzero8() {
  v8f z = {0.f,0.f,0.f,0.f,0.f,0.f,0.f,0.f};
  return z;
}

__device__ __forceinline__ v8f wmma_bf16(v16bf a, v16bf b, v8f c) {
  return __builtin_amdgcn_wmma_f32_16x16x32_bf16(false, a, false, b,
                                                 (short)0, c, false, false);
}

// A fragment: 16x32 bf16, lane L holds row M = m0 + (L&15),
// element j holds K = k0 + (j&7) + 16*(j>>3) + 8*(L>>4).
template<int CS>
__device__ __forceinline__ v16bf load_a_frag(const float* base, long rs,
                                             int m0, int k0, int lane) {
  const int mr = m0 + (lane & 15);
  const int kb = k0 + ((lane >> 4) << 3);
  const float* p = base + (long)mr * rs + (long)kb * CS;
  v16bf f;
  if constexpr (CS == 1) {
    const float4 q0 = *(const float4*)(p);
    const float4 q1 = *(const float4*)(p + 4);
    const float4 q2 = *(const float4*)(p + 16);
    const float4 q3 = *(const float4*)(p + 20);
    f[0]=(__bf16)q0.x; f[1]=(__bf16)q0.y; f[2]=(__bf16)q0.z; f[3]=(__bf16)q0.w;
    f[4]=(__bf16)q1.x; f[5]=(__bf16)q1.y; f[6]=(__bf16)q1.z; f[7]=(__bf16)q1.w;
    f[8]=(__bf16)q2.x; f[9]=(__bf16)q2.y; f[10]=(__bf16)q2.z; f[11]=(__bf16)q2.w;
    f[12]=(__bf16)q3.x; f[13]=(__bf16)q3.y; f[14]=(__bf16)q3.z; f[15]=(__bf16)q3.w;
  } else {
#pragma unroll
    for (int j = 0; j < 16; ++j) {
      const int k = (j & 7) + ((j >> 3) << 4);
      f[j] = (__bf16)p[(long)k * CS];
    }
  }
  return f;
}

// B fragment: 32x16 bf16 (W stored as W[n][k]); lane L holds col N = n0+(L&15),
// element j holds K = k0 + (j&7) + 16*(j>>3) + 8*(L>>4).
template<int CS>
__device__ __forceinline__ v16bf load_b_frag(const float* base, long rs,
                                             int n0, int k0, int lane) {
  const int nr = n0 + (lane & 15);
  const int kb = k0 + ((lane >> 4) << 3);
  const float* p = base + (long)nr * rs + (long)kb * CS;
  v16bf f;
  if constexpr (CS == 1) {
    const float4 q0 = *(const float4*)(p);
    const float4 q1 = *(const float4*)(p + 4);
    const float4 q2 = *(const float4*)(p + 16);
    const float4 q3 = *(const float4*)(p + 20);
    f[0]=(__bf16)q0.x; f[1]=(__bf16)q0.y; f[2]=(__bf16)q0.z; f[3]=(__bf16)q0.w;
    f[4]=(__bf16)q1.x; f[5]=(__bf16)q1.y; f[6]=(__bf16)q1.z; f[7]=(__bf16)q1.w;
    f[8]=(__bf16)q2.x; f[9]=(__bf16)q2.y; f[10]=(__bf16)q2.z; f[11]=(__bf16)q2.w;
    f[12]=(__bf16)q3.x; f[13]=(__bf16)q3.y; f[14]=(__bf16)q3.z; f[15]=(__bf16)q3.w;
  } else {
#pragma unroll
    for (int j = 0; j < 16; ++j) f[j] = (__bf16)p[(long)((j & 7) + ((j >> 3) << 4)) * CS];
  }
  return f;
}

__device__ __forceinline__ float wave_sum(float v) {
#pragma unroll
  for (int off = 16; off; off >>= 1) v += __shfl_xor(v, off, 32);
  return v;
}
__device__ __forceinline__ float wave_max(float v) {
#pragma unroll
  for (int off = 16; off; off >>= 1) v = fmaxf(v, __shfl_xor(v, off, 32));
  return v;
}

// ---------------------------------------------------------------------------
// Generic strided WMMA GEMM: C[z] = epilogue(A[z] @ W[z]^T)
// z-slice base = ptr + (z/zdiv)*outer + (z%zdiv)*inner.
// Block = 4 waves stacked along M; wave tile 32x64 (2x4 16x16 accumulators)
// -> 6 fragment loads per 8 WMMAs. Block tile 128x64.
// Epilogue order: +bias[n], clip(+-5), +E, ELU, +residual.
// Launch geometry must tile M(128),N(64),K(32) exactly (no bounds checks).
// ---------------------------------------------------------------------------
template<int ACS, int WCS>
__global__ __launch_bounds__(128) void wmma_gemm_kernel(
    const float* __restrict__ A, long aO, long aI, long aR,
    const float* __restrict__ W, long wO, long wI, long wR,
    float* __restrict__ Cp,      long cO, long cI, long cR, long cC,
    const float* __restrict__ bias,
    const float* __restrict__ Rres, long rO, long rI, long rR, long rC,
    const float* __restrict__ Eadd, long eO, long eI, long eR, long eC,
    int K, int zdiv, int flags)
{
  const int lane = threadIdx.x & 31;
  const int wv   = threadIdx.x >> 5;
  const int z    = blockIdx.z;
  const int zo = z / zdiv, zi = z % zdiv;
  const float* Ab = A + (long)zo * aO + (long)zi * aI;
  const float* Wb = W + (long)zo * wO + (long)zi * wI;
  float*       Cb = Cp + (long)zo * cO + (long)zi * cI;
  const int m0 = blockIdx.x * 128 + wv * 32;
  const int n0 = blockIdx.y * 64;

  v8f acc[2][4];
#pragma unroll
  for (int i = 0; i < 2; ++i)
#pragma unroll
    for (int j = 0; j < 4; ++j) acc[i][j] = vzero8();

#pragma unroll 2
  for (int k0 = 0; k0 < K; k0 += 32) {
    v16bf a0 = load_a_frag<ACS>(Ab, aR, m0,      k0, lane);
    v16bf a1 = load_a_frag<ACS>(Ab, aR, m0 + 16, k0, lane);
#pragma unroll
    for (int nt = 0; nt < 4; ++nt) {
      v16bf bf = load_b_frag<WCS>(Wb, wR, n0 + nt * 16, k0, lane);
      acc[0][nt] = wmma_bf16(a0, bf, acc[0][nt]);
      acc[1][nt] = wmma_bf16(a1, bf, acc[1][nt]);
    }
  }

  const float* Ebb = Eadd ? (Eadd + (long)zo * eO + (long)zi * eI) : nullptr;
  const float* Rbb = Rres ? (Rres + (long)zo * rO + (long)zi * rI) : nullptr;
#pragma unroll
  for (int mi = 0; mi < 2; ++mi) {
#pragma unroll
    for (int ni = 0; ni < 4; ++ni) {
      const int nn = n0 + ni * 16 + (lane & 15);
      const int mb = m0 + mi * 16 + ((lane >> 4) << 3);
#pragma unroll
      for (int r = 0; r < 8; ++r) {
        const int mm = mb + r;
        float v = acc[mi][ni][r];
        if (bias) v += bias[nn];
        if (flags & FLAG_CLIP) v = fminf(5.f, fmaxf(-5.f, v));
        if (Ebb)  v += Ebb[(long)mm * eR + (long)nn * eC];
        if (flags & FLAG_ELU) v = (v > 0.f) ? v : expm1f(v);
        if (Rbb)  v += Rbb[(long)mm * rR + (long)nn * rC];
        Cb[(long)mm * cR + (long)nn * cC] = v;
      }
    }
  }
}

// ---------------------------------------------------------------------------
// LayerNorm over dim=512, one wave per row (16 elems/lane).
// ---------------------------------------------------------------------------
__global__ __launch_bounds__(256) void ln512_kernel(
    const float* __restrict__ x, float* __restrict__ y,
    const float* __restrict__ g, const float* __restrict__ b, int rows)
{
  const int lane = threadIdx.x & 31;
  const int row  = blockIdx.x * 8 + (threadIdx.x >> 5);
  if (row >= rows) return;
  const float* xr = x + (size_t)row * 512;
  float v[16]; float s = 0.f;
#pragma unroll
  for (int j = 0; j < 16; ++j) { v[j] = xr[j * 32 + lane]; s += v[j]; }
  const float mu = wave_sum(s) * (1.f / 512.f);
  float q = 0.f;
#pragma unroll
  for (int j = 0; j < 16; ++j) { const float d = v[j] - mu; q += d * d; }
  const float inv = rsqrtf(wave_sum(q) * (1.f / 512.f) + 1e-5f);
  float* yr = y + (size_t)row * 512;
#pragma unroll
  for (int j = 0; j < 16; ++j) {
    const int c = j * 32 + lane;
    yr[c] = (v[j] - mu) * inv * g[c] + b[c];
  }
}

// ---------------------------------------------------------------------------
// Per edge: LN(e,32) -> E = e_ln@Ew^T+Eb, gates = sigmoid(e_ln@Gw^T+Gb).
// One lane per edge; outputs stored in (b,h,l,m) layout for GEMM epilogues.
// ---------------------------------------------------------------------------
__global__ __launch_bounds__(256) void edge_eg_kernel(
    const float* __restrict__ e,
    const float* __restrict__ g, const float* __restrict__ bb,
    const float* __restrict__ Ew, const float* __restrict__ Eb,
    const float* __restrict__ Gw, const float* __restrict__ Gb,
    float* __restrict__ Ews, float* __restrict__ Gates)
{
  const int edge = blockIdx.x * 256 + threadIdx.x;     // < 1048576
  const int b = edge >> 18, l = (edge >> 9) & 511, m = edge & 511;
  const float* er = e + (size_t)edge * 32;
  float v[32]; float s = 0.f;
#pragma unroll
  for (int c = 0; c < 32; ++c) { v[c] = er[c]; s += v[c]; }
  const float mu = s * (1.f / 32.f);
  float q = 0.f;
#pragma unroll
  for (int c = 0; c < 32; ++c) { const float d = v[c] - mu; q += d * d; }
  const float inv = rsqrtf(q * (1.f / 32.f) + 1e-5f);
#pragma unroll
  for (int c = 0; c < 32; ++c) v[c] = (v[c] - mu) * inv * g[c] + bb[c];
  const size_t base = (size_t)b * 2097152 + (size_t)l * 512 + (size_t)m;
#pragma unroll
  for (int o = 0; o < 8; ++o) {
    float ae = Eb[o], ag = Gb[o];
#pragma unroll
    for (int c = 0; c < 32; ++c) {
      ae += v[c] * Ew[o * 32 + c];
      ag += v[c] * Gw[o * 32 + c];
    }
    Ews  [base + (size_t)o * 262144] = ae;
    Gates[base + (size_t)o * 262144] = 1.f / (1.f + expf(-ag));
  }
}

// ---------------------------------------------------------------------------
// Softmax over m (row of 512) * gates; degrees -> scalers. One wave per
// (b,h,l) row; A_tild overwrites the gates buffer in place.
// ---------------------------------------------------------------------------
__global__ __launch_bounds__(256) void softmax_gate_kernel(
    const float* __restrict__ Hh, float* __restrict__ Gts,
    float* __restrict__ scal)
{
  const int lane = threadIdx.x & 31;
  const int r    = blockIdx.x * 8 + (threadIdx.x >> 5);  // < 16384
  const float* hr = Hh + (size_t)r * 512;
  float* gr = Gts + (size_t)r * 512;
  float x[16], p[16], g[16];
  float mx = -1e30f;
#pragma unroll
  for (int j = 0; j < 16; ++j) { x[j] = hr[j * 32 + lane]; mx = fmaxf(mx, x[j]); }
  mx = wave_max(mx);
  float s = 0.f;
#pragma unroll
  for (int j = 0; j < 16; ++j) { p[j] = expf(x[j] - mx); s += p[j]; }
  const float invs = 1.f / wave_sum(s);
  float dg = 0.f;
#pragma unroll
  for (int j = 0; j < 16; ++j) { g[j] = gr[j * 32 + lane]; dg += g[j]; }
  dg = wave_sum(dg);
#pragma unroll
  for (int j = 0; j < 16; ++j) gr[j * 32 + lane] = p[j] * invs * g[j];
  if (lane == 0) {
    const int bh = r >> 9, l = r & 511, b = bh >> 3, h = bh & 7;
    scal[(((size_t)b * 512 + l) << 3) + h] = (l < 4) ? 1.0f : log1pf(dg);
  }
}

// scale V_attn (b,l,NDIM) by scalers[(b*512+l)*8 + (c&7)]
__global__ __launch_bounds__(256) void scale_vattn_kernel(
    float* __restrict__ v, const float* __restrict__ scal)
{
  const int i = blockIdx.x * 256 + threadIdx.x;          // < 1048576
  const int row = i >> 9, h = i & 7;
  v[i] *= scal[(size_t)row * 8 + h];
}

// ---------------------------------------------------------------------------
// Fused edge output + edge FFN (all WMMA, no big intermediates):
//   e_pre = H_hat @ Oe^T + Oe_b + e ; x = LN(e_pre)
//   hid = elu(x @ W1^T + b1) ; e_out = hid @ W2^T + b2 + e_pre
// 2 waves / block, 32 edges per wave staged in LDS. LDS rows padded to a
// multiple of 4 floats so float4 fragment loads stay 16B-aligned.
// ---------------------------------------------------------------------------
__global__ __launch_bounds__(64) void edge_out_ffn_kernel(
    const float* __restrict__ Hh, const float* __restrict__ e,
    const float* __restrict__ Oew, const float* __restrict__ Oeb,
    const float* __restrict__ lg,  const float* __restrict__ lb,
    const float* __restrict__ W1,  const float* __restrict__ B1,
    const float* __restrict__ W2,  const float* __restrict__ B2,
    float* __restrict__ out)
{
  __shared__ float s_xln [2][32][36];
  __shared__ float s_hid [2][32][68];
  __shared__ float s_epre[2][32][33];
  const int lane = threadIdx.x & 31, wv = threadIdx.x >> 5;
  const int edge = blockIdx.x * 64 + wv * 32 + lane;     // exact multiple
  const int b = edge >> 18, l = (edge >> 9) & 511, m = edge & 511;
  const size_t hbase = (size_t)b * 2097152 + (size_t)l * 512 + (size_t)m;
  float hh[8];
#pragma unroll
  for (int h = 0; h < 8; ++h) hh[h] = Hh[hbase + (size_t)h * 262144];
  const float* er = e + (size_t)edge * 32;
  float pre[32]; float s = 0.f;
#pragma unroll
  for (int c = 0; c < 32; ++c) {
    float a = Oeb[c] + er[c];
#pragma unroll
    for (int h = 0; h < 8; ++h) a += hh[h] * Oew[c * 8 + h];
    pre[c] = a; s += a;
  }
  const float mu = s * (1.f / 32.f);
  float q = 0.f;
#pragma unroll
  for (int c = 0; c < 32; ++c) { const float d = pre[c] - mu; q += d * d; }
  const float inv = rsqrtf(q * (1.f / 32.f) + 1e-5f);
#pragma unroll
  for (int c = 0; c < 32; ++c) {
    s_epre[wv][lane][c] = pre[c];
    s_xln [wv][lane][c] = (pre[c] - mu) * inv * lg[c] + lb[c];
  }
  __syncthreads();

  // GEMM1: hid(32x64) = elu(xln(32x32) @ W1^T + B1)
  const float* xp = &s_xln[wv][0][0];
  v16bf xa0 = load_a_frag<1>(xp, 36, 0,  0, lane);
  v16bf xa1 = load_a_frag<1>(xp, 36, 16, 0, lane);
  v8f acc1[2][4];
#pragma unroll
  for (int i = 0; i < 2; ++i)
#pragma unroll
    for (int j = 0; j < 4; ++j) acc1[i][j] = vzero8();
#pragma unroll
  for (int nt = 0; nt < 4; ++nt) {
    v16bf bf = load_b_frag<1>(W1, 32, nt * 16, 0, lane);
    acc1[0][nt] = wmma_bf16(xa0, bf, acc1[0][nt]);
    acc1[1][nt] = wmma_bf16(xa1, bf, acc1[1][nt]);
  }
#pragma unroll
  for (int mi = 0; mi < 2; ++mi)
#pragma unroll
    for (int nt = 0; nt < 4; ++nt) {
      const int n  = nt * 16 + (lane & 15);
      const int mb = mi * 16 + ((lane >> 4) << 3);
#pragma unroll
      for (int r = 0; r < 8; ++r) {
        const float vv = acc1[mi][nt][r] + B1[n];
        s_hid[wv][mb + r][n] = (vv > 0.f) ? vv : expm1f(vv);
      }
    }
  __syncthreads();

  // GEMM2: out(32x32) = hid(32x64) @ W2^T + B2 + e_pre
  const float* hp = &s_hid[wv][0][0];
  v8f acc2[2][2];
#pragma unroll
  for (int i = 0; i < 2; ++i)
#pragma unroll
    for (int j = 0; j < 2; ++j) acc2[i][j] = vzero8();
#pragma unroll
  for (int k0 = 0; k0 < 64; k0 += 32) {
    v16bf ha0 = load_a_frag<1>(hp, 68, 0,  k0, lane);
    v16bf ha1 = load_a_frag<1>(hp, 68, 16, k0, lane);
#pragma unroll
    for (int nt = 0; nt < 2; ++nt) {
      v16bf bf = load_b_frag<1>(W2, 64, nt * 16, k0, lane);
      acc2[0][nt] = wmma_bf16(ha0, bf, acc2[0][nt]);
      acc2[1][nt] = wmma_bf16(ha1, bf, acc2[1][nt]);
    }
  }
  const size_t obase = (size_t)(blockIdx.x * 64 + wv * 32) * 32;
#pragma unroll
  for (int mi = 0; mi < 2; ++mi)
#pragma unroll
    for (int nt = 0; nt < 2; ++nt) {
      const int n  = nt * 16 + (lane & 15);
      const int mb = mi * 16 + ((lane >> 4) << 3);
#pragma unroll
      for (int r = 0; r < 8; ++r) {
        const int mm = mb + r;
        out[obase + (size_t)mm * 32 + n] =
            acc2[mi][nt][r] + B2[n] + s_epre[wv][mm][n];
      }
    }
}

// ---------------------------------------------------------------------------
// Host side
// ---------------------------------------------------------------------------
template<int ACS, int WCS>
static void launch_gemm(hipStream_t s,
    const float* A, long aO, long aI, long aR,
    const float* W, long wO, long wI, long wR,
    float* C,       long cO, long cI, long cR, long cC,
    const float* bias,
    const float* R, long rO, long rI, long rR, long rC,
    const float* E, long eO, long eI, long eR, long eC,
    int M, int N, int K, int z, int zdiv, int flags)
{
  dim3 grid((M + 127) / 128, (N + 63) / 64, z);
  wmma_gemm_kernel<ACS, WCS><<<grid, dim3(128), 0, s>>>(
      A, aO, aI, aR, W, wO, wI, wR, C, cO, cI, cR, cC,
      bias, R, rO, rI, rR, rC, E, eO, eI, eR, eC, K, zdiv, flags);
}

extern "C" void kernel_launch(void* const* d_in, const int* in_sizes, int n_in,
                              void* d_out, int out_size, void* d_ws, size_t ws_size,
                              hipStream_t stream)
{
  (void)in_sizes; (void)n_in; (void)out_size; (void)ws_size;
  const float* h_in     = (const float*)d_in[0];
  const float* e_in     = (const float*)d_in[1];
  const float* ln_h_g   = (const float*)d_in[2];
  const float* ln_h_b   = (const float*)d_in[3];
  const float* ln_e_g   = (const float*)d_in[4];
  const float* ln_e_b   = (const float*)d_in[5];
  const float* E_w      = (const float*)d_in[6];
  const float* E_b      = (const float*)d_in[7];
  const float* G_w      = (const float*)d_in[8];
  const float* G_b      = (const float*)d_in[9];
  const float* QKV_w    = (const float*)d_in[10];
  const float* QKV_b    = (const float*)d_in[11];
  const float* Oh_w     = (const float*)d_in[12];
  const float* Oh_b     = (const float*)d_in[13];
  const float* nffn_lng = (const float*)d_in[14];
  const float* nffn_lnb = (const float*)d_in[15];
  const float* nffn_w1  = (const float*)d_in[16];
  const float* nffn_b1  = (const float*)d_in[17];
  const float* nffn_w2  = (const float*)d_in[18];
  const float* nffn_b2  = (const float*)d_in[19];
  const float* Oe_w     = (const float*)d_in[20];
  const float* Oe_b     = (const float*)d_in[21];
  const float* effn_lng = (const float*)d_in[22];
  const float* effn_lnb = (const float*)d_in[23];
  const float* effn_w1  = (const float*)d_in[24];
  const float* effn_b1  = (const float*)d_in[25];
  const float* effn_w2  = (const float*)d_in[26];
  const float* effn_b2  = (const float*)d_in[27];

  float* out_h = (float*)d_out;               // (4,512,512)
  float* out_e = out_h + 1048576;             // (4,512,512,32)

  float* ws    = (float*)d_ws;
  float* qkvb  = ws;                          // 3,145,728 (qkv / ffn hidden)
  float* tmpb  = ws + 3145728;                // 1,048,576 (h_ln / V_attn / x_ln)
  float* E_ws  = ws + 4194304;                // 8,388,608 (b,h,l,m)
  float* G_ws  = ws + 12582912;               // 8,388,608 (gates -> A_tild)
  float* HH_ws = ws + 20971520;               // 8,388,608 (H_hat, (b,h,l,m))
  float* SC_ws = ws + 29360128;               // 16,384    (scalers (b,l,h))

  // 1. h_ln = LN(h)
  ln512_kernel<<<256, 256, 0, stream>>>(h_in, tmpb, ln_h_g, ln_h_b, 2048);
  // 2. edge LN + E + gates
  edge_eg_kernel<<<4096, 256, 0, stream>>>(e_in, ln_e_g, ln_e_b,
                                           E_w, E_b, G_w, G_b, E_ws, G_ws);
  // 3. qkv = h_ln @ QKV_w^T + b     (2048 x 1536, K=512)
  launch_gemm<1,1>(stream, tmpb, 0,0, 512,  QKV_w, 0,0, 512,
              qkvb, 0,0, 1536,1, QKV_b,
              nullptr,0,0,0,0, nullptr,0,0,0,0, 2048, 1536, 512, 1, 1, 0);
  // 4. H_hat = clip(Q@K^T) + E      (per (b,h): 512x512, K=64)
  launch_gemm<8,8>(stream, qkvb,       786432,1, 1536,
                      qkvb + 512, 786432,1, 1536,
                      HH_ws, 2097152,262144, 512,1, nullptr,
              nullptr,0,0,0,0,
              E_ws, 2097152,262144, 512,1,
              512, 512, 64, 32, 8, FLAG_CLIP);
  // 5. fused e_out + edge FFN -> out_e
  edge_out_ffn_kernel<<<16384, 64, 0, stream>>>(HH_ws, e_in, Oe_w, Oe_b,
      effn_lng, effn_lnb, effn_w1, effn_b1, effn_w2, effn_b2, out_e);
  // 6. A_tild = softmax(H_hat)*gates (in place over gates); scalers
  softmax_gate_kernel<<<2048, 256, 0, stream>>>(HH_ws, G_ws, SC_ws);
  // 7. V_attn = A_tild @ V          (per (b,h): 512x64, K=512) -> (b,l,d*8+h)
  launch_gemm<1,1536>(stream, G_ws, 2097152,262144, 512,
                      qkvb + 1024, 786432,1, 8,
                      tmpb, 262144,1, 512,8, nullptr,
              nullptr,0,0,0,0, nullptr,0,0,0,0, 512, 64, 512, 32, 8, 0);
  // 8. apply log1p(degree) scalers
  scale_vattn_kernel<<<4096, 256, 0, stream>>>(tmpb, SC_ws);
  // 9. h_out = V_attn @ Oh^T + b + h -> out_h
  launch_gemm<1,1>(stream, tmpb, 0,0, 512, Oh_w, 0,0, 512,
              out_h, 0,0, 512,1, Oh_b,
              h_in, 0,0, 512,1, nullptr,0,0,0,0, 2048, 512, 512, 1, 1, 0);
  // 10. x = LN(h_out)
  ln512_kernel<<<256, 256, 0, stream>>>(out_h, tmpb, nffn_lng, nffn_lnb, 2048);
  // 11. hid = elu(x @ w1^T + b1)    (2048 x 1024)
  launch_gemm<1,1>(stream, tmpb, 0,0, 512, nffn_w1, 0,0, 512,
              qkvb, 0,0, 1024,1, nffn_b1,
              nullptr,0,0,0,0, nullptr,0,0,0,0, 2048, 1024, 512, 1, 1, FLAG_ELU);
  // 12. h_out += hid @ w2^T + b2
  launch_gemm<1,1>(stream, qkvb, 0,0, 1024, nffn_w2, 0,0, 1024,
              out_h, 0,0, 512,1, nffn_b2,
              out_h, 0,0, 512,1, nullptr,0,0,0,0, 2048, 512, 1024, 1, 1, 0);
}
